// RNN_67886253080777
// MI455X (gfx1250) — matrix operand order repaired and checked by reading, not compiled, not run on hardware
//
#include <hip/hip_runtime.h>
#include <hip/hip_bf16.h>

// ---------------------------------------------------------------------------
// Types
// ---------------------------------------------------------------------------
typedef __bf16 bf16_t;
typedef __bf16 bf16x4  __attribute__((ext_vector_type(4)));
typedef __bf16 bf16x8  __attribute__((ext_vector_type(8)));
typedef __bf16 v16bf   __attribute__((ext_vector_type(16)));
typedef float  v8f     __attribute__((ext_vector_type(8)));

#define WMMA_BF16(acc, a, b)                                                  \
  (acc) = __builtin_amdgcn_wmma_f32_16x16x32_bf16(                            \
      false, (a), false, (b), (short)0, (acc), false, false)

// Problem constants (match reference)
constexpr int BB   = 32;
constexpr int NV   = 256;
constexpr int TT   = 12;
constexpr int FIN  = 64;
constexpr int HD   = 256;
constexpr int M    = BB * NV;        // 8192 sequences
constexpr int LSTR = 264;            // padded LDS row stride (bf16 elems), 528B = 33*16

// Branchless tanh on the hardware transcendental path:
//   tanh(x) = 1 - 2/(exp(2x)+1);  exp->v_exp_f32, rcp->v_rcp_f32.
// Saturates correctly: exp(2x)->inf => 1, exp(2x)->0 => -1.
__device__ __forceinline__ float fast_tanh(float x) {
  float e = __expf(2.0f * x);
  return 1.0f - 2.0f * __builtin_amdgcn_rcpf(e + 1.0f);
}

// ---------------------------------------------------------------------------
// WMMA fragment loaders (wave32, 16x16x32 bf16)
//   A (16xK, 16-bit): lanes 0-15 row M=lane, VGPR0-3 = K[k0+8hi .. +7],
//                     VGPR4-7 = K[k0+16+8hi .. +7]  -> two b128 loads
//   B (Kx16, 16-bit): lane n=lane%16, same K order as A:
//                     K[k0+8hi .. +7], K[k0+16+8hi .. +7] -> two b128 loads
// ---------------------------------------------------------------------------
__device__ __forceinline__ v16bf frag_a_ptr(const bf16_t* p, int hi) {
  const bf16x8* p0 = (const bf16x8*)(p + hi * 8);
  const bf16x8* p1 = (const bf16x8*)(p + 16 + hi * 8);
  bf16x8 a = *p0, b = *p1;
  return __builtin_shufflevector(a, b, 0, 1, 2, 3, 4, 5, 6, 7, 8, 9, 10, 11,
                                 12, 13, 14, 15);
}
__device__ __forceinline__ v16bf frag_b_ptr(const bf16_t* p, int hi) {
  const bf16x8* p0 = (const bf16x8*)(p + hi * 8);
  const bf16x8* p1 = (const bf16x8*)(p + 16 + hi * 8);
  bf16x8 a = *p0, b = *p1;
  return __builtin_shufflevector(a, b, 0, 1, 2, 3, 4, 5, 6, 7, 8, 9, 10, 11,
                                 12, 13, 14, 15);
}

// ---------------------------------------------------------------------------
// Kernel 0: f32 -> bf16 conversion (x and weights), vectorized x4
// ---------------------------------------------------------------------------
__global__ void cvt_kernel(const float* __restrict__ s, bf16_t* __restrict__ d,
                           int n4) {
  int i = blockIdx.x * 256 + threadIdx.x;
  if (i >= n4) return;
  float4 v = ((const float4*)s)[i];
  bf16x4 o;
  o[0] = (bf16_t)v.x; o[1] = (bf16_t)v.y; o[2] = (bf16_t)v.z; o[3] = (bf16_t)v.w;
  ((bf16x4*)d)[i] = o;
}

// ---------------------------------------------------------------------------
// Kernel 1: fused 2-layer RNN scan.
// Block = 16 sequences, 256 threads (8 waves). Wave w owns N-tiles {2w,2w+1}.
// Recurrent weights live in VGPRs as B-fragments; h state lives in LDS.
// LDS A-fragment loads are software-pipelined one fragment ahead so the
// ds_load wait overlaps the previous fragment's WMMAs.
// ---------------------------------------------------------------------------
__global__ __launch_bounds__(256) void rnn2_kernel(
    const bf16_t* __restrict__ xb,     // [M, T, 64]  bf16
    const bf16_t* __restrict__ wih0,   // [256, 64]
    const bf16_t* __restrict__ whh0,   // [256, 256]
    const bf16_t* __restrict__ wih1,   // [256, 256]
    const bf16_t* __restrict__ whh1,   // [256, 256]
    const float* __restrict__ bih0, const float* __restrict__ bhh0,
    const float* __restrict__ bih1, const float* __restrict__ bhh1,
    bf16_t* __restrict__ h2out)        // [M, T, 256] bf16
{
  extern __shared__ __align__(16) char smem[];
  bf16_t* sh_h  = (bf16_t*)smem;            // [16][LSTR]      current hidden
  bf16_t* sh_ht = sh_h + 16 * LSTR;         // [12][16][LSTR]  h1 -> pre1 -> h2

  const int tid  = threadIdx.x;
  const int lane = tid & 31;
  const int w    = tid >> 5;
  const int r    = lane & 15;
  const int hi   = lane >> 4;
  const int n0   = w * 32;
  const int m0   = blockIdx.x * 16;

  for (int i = tid; i < 16 * LSTR; i += 256) sh_h[i] = (bf16_t)0.0f;

  float bias0[2], bias1[2];
#pragma unroll
  for (int j = 0; j < 2; ++j) {
    int n = n0 + j * 16 + r;
    bias0[j] = bih0[n] + bhh0[n];
    bias1[j] = bih1[n] + bhh1[n];
  }

  // Register-resident B fragments: W_ih0 (K=64 -> 2 kf) and W_hh0 (8 kf)
  v16bf fih[2][2];
#pragma unroll
  for (int j = 0; j < 2; ++j)
#pragma unroll
    for (int kf = 0; kf < 2; ++kf)
      fih[j][kf] = frag_b_ptr(wih0 + (n0 + j * 16 + r) * FIN + kf * 32, hi);

  v16bf fhh[2][8];
#pragma unroll
  for (int j = 0; j < 2; ++j)
#pragma unroll
    for (int kf = 0; kf < 8; ++kf)
      fhh[j][kf] = frag_b_ptr(whh0 + (n0 + j * 16 + r) * HD + kf * 32, hi);

  __syncthreads();

  // ---------------- Layer 0 scan:  h = tanh(x W_ih0^T + b + h W_hh0^T) -----
  const bf16_t* xrow = xb + ((size_t)(m0 + r) * TT) * FIN;
  // prefetch first timestep's x fragments
  v16bf ax0 = frag_a_ptr(xrow + 0, hi);
  v16bf ax1 = frag_a_ptr(xrow + 32, hi);
  for (int t = 0; t < TT; ++t) {
    v8f acc[2];
#pragma unroll
    for (int j = 0; j < 2; ++j)
#pragma unroll
      for (int i = 0; i < 8; ++i) acc[j][i] = bias0[j];

    WMMA_BF16(acc[0], ax0, fih[0][0]);
    WMMA_BF16(acc[1], ax0, fih[1][0]);
    WMMA_BF16(acc[0], ax1, fih[0][1]);
    WMMA_BF16(acc[1], ax1, fih[1][1]);
    // prefetch next timestep's x fragments during the recurrent WMMAs
    if (t + 1 < TT) {
      ax0 = frag_a_ptr(xrow + (t + 1) * FIN + 0, hi);
      ax1 = frag_a_ptr(xrow + (t + 1) * FIN + 32, hi);
    }

    // recurrent part, pipelined one LDS fragment ahead
    v16bf a0 = frag_a_ptr(sh_h + r * LSTR, hi);
#pragma unroll
    for (int kf = 0; kf < 8; ++kf) {
      v16bf a1 = frag_a_ptr(sh_h + r * LSTR + (((kf + 1) & 7) * 32), hi);
      WMMA_BF16(acc[0], a0, fhh[0][kf]);
      WMMA_BF16(acc[1], a0, fhh[1][kf]);
      a0 = a1;
    }
    __syncthreads();  // all reads of old h complete
    bf16_t* dst = sh_ht + t * (16 * LSTR);
#pragma unroll
    for (int j = 0; j < 2; ++j)
#pragma unroll
      for (int i = 0; i < 8; ++i) {
        bf16_t v = (bf16_t)fast_tanh(acc[j][i]);
        int idx  = (i + 8 * hi) * LSTR + n0 + j * 16 + r;
        sh_h[idx] = v;   // new hidden state
        dst[idx]  = v;   // keep h1[t] for layer 1
      }
    __syncthreads();
  }

  // ---------------- pre1[t] = h1[t] W_ih1^T + b1  (in place in sh_ht) ------
  v16bf f1[2][8];
#pragma unroll
  for (int j = 0; j < 2; ++j)
#pragma unroll
    for (int kf = 0; kf < 8; ++kf)
      f1[j][kf] = frag_b_ptr(wih1 + (n0 + j * 16 + r) * HD + kf * 32, hi);

  for (int t = 0; t < TT; ++t) {
    bf16_t* buf = sh_ht + t * (16 * LSTR);
    v8f acc[2];
#pragma unroll
    for (int j = 0; j < 2; ++j)
#pragma unroll
      for (int i = 0; i < 8; ++i) acc[j][i] = bias1[j];
    v16bf a0 = frag_a_ptr(buf + r * LSTR, hi);
#pragma unroll
    for (int kf = 0; kf < 8; ++kf) {
      v16bf a1 = frag_a_ptr(buf + r * LSTR + (((kf + 1) & 7) * 32), hi);
      WMMA_BF16(acc[0], a0, f1[0][kf]);
      WMMA_BF16(acc[1], a0, f1[1][kf]);
      a0 = a1;
    }
    __syncthreads();
#pragma unroll
    for (int j = 0; j < 2; ++j)
#pragma unroll
      for (int i = 0; i < 8; ++i)
        buf[(i + 8 * hi) * LSTR + n0 + j * 16 + r] = (bf16_t)acc[j][i];
    __syncthreads();
  }

  // Reload frags with W_hh1; reset hidden state
#pragma unroll
  for (int j = 0; j < 2; ++j)
#pragma unroll
    for (int kf = 0; kf < 8; ++kf)
      f1[j][kf] = frag_b_ptr(whh1 + (n0 + j * 16 + r) * HD + kf * 32, hi);
  for (int i = tid; i < 16 * LSTR; i += 256) sh_h[i] = (bf16_t)0.0f;
  __syncthreads();

  // ---------------- Layer 1 scan: h = tanh(pre1[t] + h W_hh1^T) ------------
  for (int t = 0; t < TT; ++t) {
    bf16_t* buf = sh_ht + t * (16 * LSTR);
    v8f acc[2];
#pragma unroll
    for (int j = 0; j < 2; ++j)
#pragma unroll
      for (int i = 0; i < 8; ++i)
        acc[j][i] = (float)buf[(i + 8 * hi) * LSTR + n0 + j * 16 + r];
    v16bf a0 = frag_a_ptr(sh_h + r * LSTR, hi);
#pragma unroll
    for (int kf = 0; kf < 8; ++kf) {
      v16bf a1 = frag_a_ptr(sh_h + r * LSTR + (((kf + 1) & 7) * 32), hi);
      WMMA_BF16(acc[0], a0, f1[0][kf]);
      WMMA_BF16(acc[1], a0, f1[1][kf]);
      a0 = a1;
    }
    __syncthreads();
#pragma unroll
    for (int j = 0; j < 2; ++j)
#pragma unroll
      for (int i = 0; i < 8; ++i) {
        bf16_t v = (bf16_t)fast_tanh(acc[j][i]);
        int idx  = (i + 8 * hi) * LSTR + n0 + j * 16 + r;
        sh_h[idx] = v;
        buf[idx]  = v;   // slot now holds h2[t]
      }
    __syncthreads();
  }

  // ---------------- write h2 tile -> global (vector b128 stores) -----------
  for (int q = 0; q < 24; ++q) {
    int e   = (q * 256 + tid) * 8;       // element index in 12*16*256
    int t   = e >> 12;                   // / 4096
    int rem = e & 4095;
    int mm  = rem >> 8;
    int h   = rem & 255;
    bf16x8 v = *(const bf16x8*)(sh_ht + (t * 16 + mm) * LSTR + h);
    *(bf16x8*)(h2out + ((size_t)(m0 + mm) * TT + t) * HD + h) = v;
  }
}

// ---------------------------------------------------------------------------
// Kernel 2: ys = time-mix(h2) + b_seq (VALU, into LDS as bf16),
//           y  = ys W_feat^T + b_feat  (WMMA, f32 out)
// Block = 16 sequences -> 192 output rows.
// ---------------------------------------------------------------------------
__global__ __launch_bounds__(256) void head_kernel(
    const bf16_t* __restrict__ h2,     // [M, T, 256] bf16
    const float* __restrict__ wseq,    // [12, 12]
    const float* __restrict__ bseq,    // [12]
    const bf16_t* __restrict__ wfeat,  // [64, 256] bf16
    const float* __restrict__ bfeat,   // [64]
    float* __restrict__ out)           // [M, 12, 64] f32
{
  extern __shared__ __align__(16) char smem[];
  bf16_t* sh_ys = (bf16_t*)smem;       // [192][LSTR]
  __shared__ float s_wseq[144];
  __shared__ float s_bseq[12];

  const int tid = threadIdx.x;
  if (tid < 144) s_wseq[tid] = wseq[tid];
  if (tid < 12)  s_bseq[tid] = bseq[tid];
  __syncthreads();

  const int m0 = blockIdx.x * 16;

  // Phase 1: ys[r, s, h] = sum_t wseq[s,t] * h2[m0+r, t, h] + bseq[s]
  {
    const int rr = tid >> 4;                 // local sequence 0..15
    const int hb = (tid & 15) << 4;          // 16 h-columns per thread
    const bf16_t* hrow = h2 + (size_t)(m0 + rr) * TT * HD + hb;
    for (int c = 0; c < 2; ++c) {            // two chunks of 8 h each
      float acc[12][8];
#pragma unroll
      for (int s = 0; s < 12; ++s)
#pragma unroll
        for (int j = 0; j < 8; ++j) acc[s][j] = 0.0f;
      for (int t = 0; t < TT; ++t) {
        bf16x8 v = *(const bf16x8*)(hrow + t * HD + c * 8);
        float f[8];
#pragma unroll
        for (int j = 0; j < 8; ++j) f[j] = (float)v[j];
#pragma unroll
        for (int s = 0; s < 12; ++s) {
          float wv = s_wseq[s * 12 + t];
#pragma unroll
          for (int j = 0; j < 8; ++j) acc[s][j] += wv * f[j];
        }
      }
#pragma unroll
      for (int s = 0; s < 12; ++s) {
        float bs = s_bseq[s];
        bf16x8 o;
#pragma unroll
        for (int j = 0; j < 8; ++j) o[j] = (bf16_t)(acc[s][j] + bs);
        *(bf16x8*)(sh_ys + (rr * 12 + s) * LSTR + hb + c * 8) = o;
      }
    }
  }
  __syncthreads();

  // Phase 2: [192,256] x W_feat^T -> [192,64] via WMMA; b_feat in acc init
  const int lane = tid & 31;
  const int w    = tid >> 5;
  const int r    = lane & 15;
  const int hi   = lane >> 4;
  for (int q = 0; q < 6; ++q) {              // 48 tiles / 8 waves
    int idx = w * 6 + q;
    int mt  = idx >> 2;                      // 0..11
    int nt  = idx & 3;                       // 0..3
    float bf = bfeat[nt * 16 + r];
    v8f acc;
#pragma unroll
    for (int i = 0; i < 8; ++i) acc[i] = bf;
    v16bf a0 = frag_a_ptr(sh_ys + (mt * 16 + r) * LSTR, hi);
#pragma unroll
    for (int kf = 0; kf < 8; ++kf) {
      v16bf a1 = frag_a_ptr(sh_ys + (mt * 16 + r) * LSTR + (((kf + 1) & 7) * 32), hi);
      v16bf b = frag_b_ptr(wfeat + (nt * 16 + r) * HD + kf * 32, hi);
      WMMA_BF16(acc, a0, b);
      a0 = a1;
    }
#pragma unroll
    for (int i = 0; i < 8; ++i) {
      int row = mt * 16 + i + 8 * hi;
      out[((size_t)m0 * TT + row) * FIN + nt * 16 + r] = acc[i];
    }
  }
}

// ---------------------------------------------------------------------------
// Host launcher
// ---------------------------------------------------------------------------
extern "C" void kernel_launch(void* const* d_in, const int* in_sizes, int n_in,
                              void* d_out, int out_size, void* d_ws,
                              size_t ws_size, hipStream_t stream) {
  const float* x     = (const float*)d_in[0];
  const float* Wih0  = (const float*)d_in[1];
  const float* Whh0  = (const float*)d_in[2];
  const float* bih0  = (const float*)d_in[3];
  const float* bhh0  = (const float*)d_in[4];
  const float* Wih1  = (const float*)d_in[5];
  const float* Whh1  = (const float*)d_in[6];
  const float* bih1  = (const float*)d_in[7];
  const float* bhh1  = (const float*)d_in[8];
  const float* Wseq  = (const float*)d_in[9];
  const float* bseq  = (const float*)d_in[10];
  const float* Wfeat = (const float*)d_in[11];
  const float* bfeat = (const float*)d_in[12];
  float* out = (float*)d_out;

  char* ws = (char*)d_ws;
  size_t off = 0;
  bf16_t* xb     = (bf16_t*)(ws + off); off += (size_t)M * TT * FIN * 2;
  bf16_t* wih0b  = (bf16_t*)(ws + off); off += (size_t)HD * FIN * 2;
  bf16_t* whh0b  = (bf16_t*)(ws + off); off += (size_t)HD * HD * 2;
  bf16_t* wih1b  = (bf16_t*)(ws + off); off += (size_t)HD * HD * 2;
  bf16_t* whh1b  = (bf16_t*)(ws + off); off += (size_t)HD * HD * 2;
  bf16_t* wfeatb = (bf16_t*)(ws + off); off += (size_t)FIN * HD * 2;
  bf16_t* h2b    = (bf16_t*)(ws + off); off += (size_t)M * TT * HD * 2;

  auto cvt = [&](const float* s, bf16_t* d, int n) {
    int n4 = n / 4;
    cvt_kernel<<<(n4 + 255) / 256, 256, 0, stream>>>(s, d, n4);
  };
  cvt(x, xb, M * TT * FIN);
  cvt(Wih0, wih0b, HD * FIN);
  cvt(Whh0, whh0b, HD * HD);
  cvt(Wih1, wih1b, HD * HD);
  cvt(Whh1, whh1b, HD * HD);
  cvt(Wfeat, wfeatb, FIN * HD);

  const int SMEM1 = (16 * LSTR + TT * 16 * LSTR) * (int)sizeof(bf16_t);
  const int SMEM2 = 192 * LSTR * (int)sizeof(bf16_t);
  hipFuncSetAttribute(reinterpret_cast<const void*>(rnn2_kernel),
                      hipFuncAttributeMaxDynamicSharedMemorySize, SMEM1);
  hipFuncSetAttribute(reinterpret_cast<const void*>(head_kernel),
                      hipFuncAttributeMaxDynamicSharedMemorySize, SMEM2);

  rnn2_kernel<<<M / 16, 256, SMEM1, stream>>>(xb, wih0b, whh0b, wih1b, whh1b,
                                              bih0, bhh0, bih1, bhh1, h2b);
  head_kernel<<<M / 16, 256, SMEM2, stream>>>(h2b, Wseq, bseq, wfeatb, bfeat,
                                              out);
  (void)in_sizes; (void)n_in; (void)out_size; (void)ws_size;
}
